// CRaWl_33036888441429
// MI455X (gfx1250) — matrix operand order repaired and checked by reading, not compiled, not run on hardware
//
#include <hip/hip_runtime.h>
#include <hip/hip_bf16.h>
#include <stdint.h>

// ---------------- problem constants (match reference) ----------------
#define NN   50000   // nodes
#define EEC  400000  // edges
#define WWC  10000   // walks
#define LLC  50      // walk length
#define KKC  9       // depthwise kernel
#define DNF  64      // node feat dim (layer0 input)
#define DEF  32      // edge feat dim
#define DWF  16      // walk struct feat dim
#define HHC  128     // hidden
#define CDIM 128     // conv dim
#define BBC  128     // graphs
#define OUTD 8
#define EPSC 1e-5f

// ---------------- WMMA types & fragment loaders ----------------
typedef __attribute__((ext_vector_type(16))) _Float16 v16h;
typedef __attribute__((ext_vector_type(8)))  float    v8f;

union AB16 { v16h v; uint4 q[2]; };

// A-matrix (16x32 f16) tile from row-major [rows][ld] at (m0,k0).
// lanes 0-15: row=lane,    K = k0+0..7  and k0+16..23
// lanes16-31: row=lane-16, K = k0+8..15 and k0+24..31
__device__ __forceinline__ v16h load_a16(const _Float16* Wp, int ld, int m0, int k0, int lane) {
  int row = m0 + (lane & 15);
  int kb  = k0 + ((lane >> 4) << 3);
  const _Float16* p = Wp + (size_t)row * ld + kb;
  AB16 u;
  u.q[0] = *reinterpret_cast<const uint4*>(p);
  u.q[1] = *reinterpret_cast<const uint4*>(p + 16);
  return u.v;
}

// B-matrix (32x16 f16) tile: column n contiguous-in-k storage [ncols][ldk].
// lane holds col = n0+(lane&15), K = k0+8h+0..7 and k0+16+8h+0..7 (h = lane>=16)
__device__ __forceinline__ v16h load_b16(const _Float16* Bp, int ldk, int k0, int n0, int lane) {
  int col = n0 + (lane & 15);
  int kb  = k0 + ((lane >> 4) << 3);
  const _Float16* p = Bp + (size_t)col * ldk + kb;
  AB16 u;
  u.q[0] = *reinterpret_cast<const uint4*>(p);
  u.q[1] = *reinterpret_cast<const uint4*>(p + 16);
  return u.v;
}

__device__ __forceinline__ v8f wmma16(v16h a, v16h b, v8f c) {
  return __builtin_amdgcn_wmma_f32_16x16x32_f16(false, a, false, b, (short)0, c, false, false);
}

__device__ __forceinline__ v8f vzero8() {
  v8f z;
#pragma unroll
  for (int r = 0; r < 8; ++r) z[r] = 0.0f;
  return z;
}

// ---------------- prep: f32 -> f16 with column zero-pad ----------------
__global__ void cvt_pad_kernel(const float* __restrict__ src, _Float16* __restrict__ dst,
                               int rows, int csrc, int cdst) {
  int idx = blockIdx.x * blockDim.x + threadIdx.x;
  if (idx >= rows * cdst) return;
  int r = idx / cdst, c = idx - r * cdst;
  dst[idx] = (c < csrc) ? (_Float16)src[(size_t)r * csrc + c] : (_Float16)0.0f;
}

// ---------------- fused walk layer: gather + conv1(WMMA) + depthwise + BN +
//                  ReLU + conv2(WMMA) + ReLU + scatter-mean accumulation ----
template <int DN_, int KPAD>
__global__ __launch_bounds__(256) void walk_layer_kernel(
    const float* __restrict__ hsrc,        // [N][DN_]
    const float* __restrict__ edge_attr,   // [E][32]
    const int*   __restrict__ walk_nodes,  // [W][50]
    const int*   __restrict__ walk_edges,  // [W][49]
    const float* __restrict__ walk_x,      // [W][16][50]
    const _Float16* __restrict__ w1,       // [128][KPAD] f16 (conv1, K-padded)
    const _Float16* __restrict__ w2,       // [128][128]  f16 (conv2)
    const float* __restrict__ dw,          // [128][9]
    const float* __restrict__ bn_g, const float* __restrict__ bn_b,
    const float* __restrict__ bn_m, const float* __restrict__ bn_v,
    float* __restrict__ p_sum, float* __restrict__ p_cnt) {
  __shared__ __align__(16) union {
    _Float16 xc[64 * KPAD];   // [pos][k] column(pos)-major, k contiguous
    float    y1[128 * 64];    // conv1 result [c][pos]
  } u;
  __shared__ __align__(16) _Float16 y2[48 * 128];  // [pos][c], c contiguous
  __shared__ int   nodes[64];
  __shared__ float bninv[128], bnadd[128];

  const int w    = blockIdx.x;
  const int tid  = threadIdx.x;
  const int lane = tid & 31;
  const int wave = tid >> 5;
  const int m0   = wave * 16;

  if (tid < LLC) nodes[tid] = walk_nodes[w * LLC + tid];
  if (tid < CDIM) {
    float inv = bn_g[tid] * rsqrtf(bn_v[tid] + EPSC);
    bninv[tid] = inv;
    bnadd[tid] = bn_b[tid] - bn_m[tid] * inv;
  }
  __syncthreads();

  // gather xc = concat(h[walk_nodes], walk_edge_h(front-padded), walk_x)
  constexpr int CIN = DN_ + DEF + DWF;
  for (int e = tid; e < 64 * KPAD; e += 256) {
    int pos = e / KPAD, k = e - pos * KPAD;
    float val = 0.0f;
    if (pos < LLC && k < CIN) {
      if (k < DN_) {
        val = hsrc[(size_t)nodes[pos] * DN_ + k];
      } else if (k < DN_ + DEF) {
        if (pos > 0)
          val = edge_attr[(size_t)walk_edges[w * (LLC - 1) + pos - 1] * DEF + (k - DN_)];
      } else {
        val = walk_x[((size_t)w * DWF + (k - DN_ - DEF)) * LLC + pos];
      }
    }
    u.xc[pos * KPAD + k] = (_Float16)val;
  }
  __syncthreads();

  // conv1: y1[128 x 50] = W1[128 x CIN] * xc  — each wave: one 16-row M tile
  v8f acc[4];
#pragma unroll
  for (int nt = 0; nt < 4; ++nt) acc[nt] = vzero8();
#pragma unroll
  for (int kt = 0; kt < KPAD / 32; ++kt) {
    v16h a = load_a16(w1, KPAD, m0, kt * 32, lane);
#pragma unroll
    for (int nt = 0; nt < 4; ++nt) {
      v16h b = load_b16(&u.xc[0], KPAD, kt * 32, nt * 16, lane);
      acc[nt] = wmma16(a, b, acc[nt]);
    }
  }
  __syncthreads();  // all waves done reading xc before overlaying y1
#pragma unroll
  for (int nt = 0; nt < 4; ++nt)
#pragma unroll
    for (int r = 0; r < 8; ++r) {
      int mm = m0 + r + ((lane >> 4) << 3);
      int nn = nt * 16 + (lane & 15);
      u.y1[mm * 64 + nn] = acc[nt][r];
    }
  __syncthreads();

  // depthwise K=9 VALID + BN1 + ReLU  -> y2 (f16, pos-major, zero-pad pos>=42)
  for (int e = tid; e < 48 * 128; e += 256) {
    int j = e >> 7, c = e & 127;
    float s = 0.0f;
    if (j < LLC - KKC + 1) {
      const float* row = &u.y1[c * 64 + j];
      const float* dwc = dw + c * KKC;
#pragma unroll
      for (int k = 0; k < KKC; ++k) s += row[k] * dwc[k];
      s = fmaxf(s * bninv[c] + bnadd[c], 0.0f);
    }
    y2[j * 128 + c] = (_Float16)s;
  }
  __syncthreads();

  // conv2: [128 x 42] = W2[128x128] * y2 , then ReLU + scatter-mean atomics
  v8f acc2[3];
#pragma unroll
  for (int nt = 0; nt < 3; ++nt) acc2[nt] = vzero8();
#pragma unroll
  for (int kt = 0; kt < 4; ++kt) {
    v16h a = load_a16(w2, 128, m0, kt * 32, lane);
#pragma unroll
    for (int nt = 0; nt < 3; ++nt) {
      v16h b = load_b16(y2, 128, kt * 32, nt * 16, lane);
      acc2[nt] = wmma16(a, b, acc2[nt]);
    }
  }
#pragma unroll
  for (int nt = 0; nt < 3; ++nt) {
    int j = nt * 16 + (lane & 15);
    if (j < LLC - KKC + 1) {
      int node = nodes[4 + j];  // pool_node = K//2 = 4
      float* dst = p_sum + (size_t)node * CDIM + m0 + ((lane >> 4) << 3);
#pragma unroll
      for (int r = 0; r < 8; ++r) atomicAdd(dst + r, fmaxf(acc2[nt][r], 0.0f));
      if (wave == 0 && lane < 16) atomicAdd(p_cnt + node, 1.0f);
    }
  }
}

// ---------------- p_node = p_sum / max(cnt,1)  -> f16 ----------------
__global__ void normalize_p_kernel(const float* __restrict__ p_sum,
                                   const float* __restrict__ p_cnt,
                                   _Float16* __restrict__ p16, int total) {
  int idx = blockIdx.x * blockDim.x + threadIdx.x;
  if (idx >= total) return;
  int node = idx >> 7;
  float c = fmaxf(p_cnt[node], 1.0f);
  p16[idx] = (_Float16)(p_sum[idx] / c);
}

// ---------------- z1 = relu(bn2(p16 @ no_w1^T))  [N x 256], f16 out ----------
__global__ __launch_bounds__(256) void node_gemm1_kernel(
    const _Float16* __restrict__ p16, const _Float16* __restrict__ w1,  // [256][128]
    const float* __restrict__ bg, const float* __restrict__ bb,
    const float* __restrict__ bm, const float* __restrict__ bv,
    _Float16* __restrict__ z16, int ntiles) {
  int lane = threadIdx.x & 31;
  int tile = blockIdx.x * 8 + (threadIdx.x >> 5);
  if (tile >= ntiles) return;
  int r0 = tile * 16;
  for (int ot = 0; ot < 16; ++ot) {
    v8f acc = vzero8();
#pragma unroll
    for (int kt = 0; kt < 4; ++kt) {
      v16h a = load_a16(p16, 128, r0, kt * 32, lane);
      v16h b = load_b16(w1, 128, kt * 32, ot * 16, lane);
      acc = wmma16(a, b, acc);
    }
    int o = ot * 16 + (lane & 15);
    float inv = bg[o] * rsqrtf(bv[o] + EPSC);
    float add = bb[o] - bm[o] * inv;
#pragma unroll
    for (int r = 0; r < 8; ++r) {
      int n = r0 + r + ((lane >> 4) << 3);
      z16[(size_t)n * 256 + o] = (_Float16)fmaxf(acc[r] * inv + add, 0.0f);
    }
  }
}

// ---------------- h = residual + z16 @ no_w2^T  (+ optional x@rescale^T) ----
__global__ __launch_bounds__(256) void node_gemm2_kernel(
    const _Float16* __restrict__ z16, const _Float16* __restrict__ w2,  // [128][256]
    const _Float16* __restrict__ x16, const _Float16* __restrict__ resc,  // [128][64] or null
    const float* __restrict__ hres,  // [N][128] or null
    float* __restrict__ hout, int ntiles, int use_resc) {
  int lane = threadIdx.x & 31;
  int tile = blockIdx.x * 8 + (threadIdx.x >> 5);
  if (tile >= ntiles) return;
  int r0 = tile * 16;
  for (int ot = 0; ot < 8; ++ot) {
    v8f acc = vzero8();
#pragma unroll
    for (int kt = 0; kt < 8; ++kt) {
      v16h a = load_a16(z16, 256, r0, kt * 32, lane);
      v16h b = load_b16(w2, 256, kt * 32, ot * 16, lane);
      acc = wmma16(a, b, acc);
    }
    if (use_resc) {
#pragma unroll
      for (int kt = 0; kt < 2; ++kt) {
        v16h a = load_a16(x16, 64, r0, kt * 32, lane);
        v16h b = load_b16(resc, 64, kt * 32, ot * 16, lane);
        acc = wmma16(a, b, acc);
      }
    }
    int o = ot * 16 + (lane & 15);
#pragma unroll
    for (int r = 0; r < 8; ++r) {
      int n = r0 + r + ((lane >> 4) << 3);
      float val = acc[r];
      if (hres) val += hres[(size_t)n * 128 + o];
      hout[(size_t)n * 128 + o] = val;
    }
  }
}

// ---------------- top BN + ReLU + graph mean pool (atomics) ----------------
__global__ void top_pool_kernel(const float* __restrict__ h, const int* __restrict__ batch,
                                const float* __restrict__ tg, const float* __restrict__ tb,
                                const float* __restrict__ tm, const float* __restrict__ tv,
                                float* __restrict__ g_sum, float* __restrict__ g_cnt, int total) {
  int idx = blockIdx.x * blockDim.x + threadIdx.x;
  if (idx >= total) return;
  int node = idx >> 7, c = idx & 127;
  float inv = tg[c] * rsqrtf(tv[c] + EPSC);
  float val = fmaxf(h[idx] * inv + (tb[c] - tm[c] * inv), 0.0f);
  int bg = batch[node];
  atomicAdd(&g_sum[bg * 128 + c], val);
  if (c == 0) atomicAdd(&g_cnt[bg], 1.0f);
}

// ---------------- final head: [B,OUT] = mean(h_graph) @ go_w^T + go_b -------
__global__ void head_kernel(const float* __restrict__ g_sum, const float* __restrict__ g_cnt,
                            const float* __restrict__ go_w, const float* __restrict__ go_b,
                            float* __restrict__ out) {
  int idx = blockIdx.x * blockDim.x + threadIdx.x;
  if (idx >= BBC * OUTD) return;
  int bg = idx >> 3, o = idx & 7;
  float cinv = 1.0f / fmaxf(g_cnt[bg], 1.0f);
  float s = 0.0f;
  for (int c = 0; c < 128; ++c) s += g_sum[bg * 128 + c] * go_w[o * 128 + c];
  out[idx] = s * cinv + go_b[o];
}

// ---------------- host launch ----------------
extern "C" void kernel_launch(void* const* d_in, const int* in_sizes, int n_in,
                              void* d_out, int out_size, void* d_ws, size_t ws_size,
                              hipStream_t stream) {
  (void)in_sizes; (void)n_in; (void)out_size; (void)ws_size;

  // positional inputs (setup_inputs dict order)
  const float* x         = (const float*)d_in[0];  // [N][64]
  const float* edge_attr = (const float*)d_in[1];  // [E][32]
  const int*   walk_nodes= (const int*)  d_in[2];  // [W][50]
  const int*   walk_edges= (const int*)  d_in[3];  // [W][49]
  const float* walk_x    = (const float*)d_in[4];  // [W][16][50]
  const int*   batch     = (const int*)  d_in[5];  // [N]
  // params pytree (sorted keys; layers list in order; layer0 has extra 'rescale')
  const float* go_b  = (const float*)d_in[6];
  const float* go_w  = (const float*)d_in[7];
  const float* l0_bn1_b = (const float*)d_in[8];
  const float* l0_bn1_g = (const float*)d_in[9];
  const float* l0_bn1_m = (const float*)d_in[10];
  const float* l0_bn1_v = (const float*)d_in[11];
  const float* l0_bn2_b = (const float*)d_in[12];
  const float* l0_bn2_g = (const float*)d_in[13];
  const float* l0_bn2_m = (const float*)d_in[14];
  const float* l0_bn2_v = (const float*)d_in[15];
  const float* l0_conv1 = (const float*)d_in[16];  // [128][112]
  const float* l0_conv2 = (const float*)d_in[17];  // [128][128]
  const float* l0_dw    = (const float*)d_in[18];  // [128][9]
  const float* l0_no_w1 = (const float*)d_in[19];  // [256][128]
  const float* l0_no_w2 = (const float*)d_in[20];  // [128][256]
  const float* l0_resc  = (const float*)d_in[21];  // [128][64]
  const float* l1_bn1_b = (const float*)d_in[22];
  const float* l1_bn1_g = (const float*)d_in[23];
  const float* l1_bn1_m = (const float*)d_in[24];
  const float* l1_bn1_v = (const float*)d_in[25];
  const float* l1_bn2_b = (const float*)d_in[26];
  const float* l1_bn2_g = (const float*)d_in[27];
  const float* l1_bn2_m = (const float*)d_in[28];
  const float* l1_bn2_v = (const float*)d_in[29];
  const float* l1_conv1 = (const float*)d_in[30];  // [128][176]
  const float* l1_conv2 = (const float*)d_in[31];
  const float* l1_dw    = (const float*)d_in[32];
  const float* l1_no_w1 = (const float*)d_in[33];
  const float* l1_no_w2 = (const float*)d_in[34];
  const float* top_b = (const float*)d_in[35];
  const float* top_g = (const float*)d_in[36];
  const float* top_m = (const float*)d_in[37];
  const float* top_v = (const float*)d_in[38];

  // workspace carve-up
  char* ws = (char*)d_ws;
  size_t off = 0;
  auto carve = [&](size_t bytes) { size_t o = off; off = (off + bytes + 255) & ~(size_t)255; return o; };
  float*    h      = (float*)   (ws + carve((size_t)NN * 128 * 4));
  float*    p_sum  = (float*)   (ws + carve((size_t)NN * 128 * 4));
  float*    p_cnt  = (float*)   (ws + carve((size_t)NN * 4));
  _Float16* p16    = (_Float16*)(ws + carve((size_t)NN * 128 * 2));
  _Float16* z16    = (_Float16*)(ws + carve((size_t)NN * 256 * 2));
  _Float16* x16    = (_Float16*)(ws + carve((size_t)NN * 64 * 2));
  _Float16* c1l0   = (_Float16*)(ws + carve(128 * 128 * 2));
  _Float16* c1l1   = (_Float16*)(ws + carve(128 * 192 * 2));
  _Float16* c2l0   = (_Float16*)(ws + carve(128 * 128 * 2));
  _Float16* c2l1   = (_Float16*)(ws + carve(128 * 128 * 2));
  _Float16* w1l0   = (_Float16*)(ws + carve(256 * 128 * 2));
  _Float16* w1l1   = (_Float16*)(ws + carve(256 * 128 * 2));
  _Float16* w2l0   = (_Float16*)(ws + carve(128 * 256 * 2));
  _Float16* w2l1   = (_Float16*)(ws + carve(128 * 256 * 2));
  _Float16* resc16 = (_Float16*)(ws + carve(128 * 64 * 2));
  float*    g_sum  = (float*)   (ws + carve(BBC * 128 * 4));
  float*    g_cnt  = (float*)   (ws + carve(BBC * 4));

  auto cvt = [&](const float* s, _Float16* d, int rows, int cs, int cd) {
    int tot = rows * cd;
    cvt_pad_kernel<<<(tot + 255) / 256, 256, 0, stream>>>(s, d, rows, cs, cd);
  };
  cvt(l0_conv1, c1l0, 128, 112, 128);
  cvt(l1_conv1, c1l1, 128, 176, 192);
  cvt(l0_conv2, c2l0, 128, 128, 128);
  cvt(l1_conv2, c2l1, 128, 128, 128);
  cvt(l0_no_w1, w1l0, 256, 128, 128);
  cvt(l1_no_w1, w1l1, 256, 128, 128);
  cvt(l0_no_w2, w2l0, 128, 256, 256);
  cvt(l1_no_w2, w2l1, 128, 256, 256);
  cvt(l0_resc,  resc16, 128, 64, 64);
  cvt(x,        x16, NN, 64, 64);

  const int ntiles = NN / 16;                // 3125 exact
  const int gblocks = (ntiles + 7) / 8;      // 8 waves per block
  const int ntot = NN * 128;

  // ---- layer 0 ----
  hipMemsetAsync(p_sum, 0, (size_t)NN * 128 * 4, stream);
  hipMemsetAsync(p_cnt, 0, (size_t)NN * 4, stream);
  walk_layer_kernel<64, 128><<<WWC, 256, 0, stream>>>(
      x, edge_attr, walk_nodes, walk_edges, walk_x, c1l0, c2l0, l0_dw,
      l0_bn1_g, l0_bn1_b, l0_bn1_m, l0_bn1_v, p_sum, p_cnt);
  normalize_p_kernel<<<(ntot + 255) / 256, 256, 0, stream>>>(p_sum, p_cnt, p16, ntot);
  node_gemm1_kernel<<<gblocks, 256, 0, stream>>>(
      p16, w1l0, l0_bn2_g, l0_bn2_b, l0_bn2_m, l0_bn2_v, z16, ntiles);
  node_gemm2_kernel<<<gblocks, 256, 0, stream>>>(
      z16, w2l0, x16, resc16, nullptr, h, ntiles, 1);

  // ---- layer 1 ----
  hipMemsetAsync(p_sum, 0, (size_t)NN * 128 * 4, stream);
  hipMemsetAsync(p_cnt, 0, (size_t)NN * 4, stream);
  walk_layer_kernel<128, 192><<<WWC, 256, 0, stream>>>(
      h, edge_attr, walk_nodes, walk_edges, walk_x, c1l1, c2l1, l1_dw,
      l1_bn1_g, l1_bn1_b, l1_bn1_m, l1_bn1_v, p_sum, p_cnt);
  normalize_p_kernel<<<(ntot + 255) / 256, 256, 0, stream>>>(p_sum, p_cnt, p16, ntot);
  node_gemm1_kernel<<<gblocks, 256, 0, stream>>>(
      p16, w1l1, l1_bn2_g, l1_bn2_b, l1_bn2_m, l1_bn2_v, z16, ntiles);
  node_gemm2_kernel<<<gblocks, 256, 0, stream>>>(
      z16, w2l1, nullptr, nullptr, h, h, ntiles, 0);

  // ---- top BN + graph pool + head ----
  hipMemsetAsync(g_sum, 0, BBC * 128 * 4, stream);
  hipMemsetAsync(g_cnt, 0, BBC * 4, stream);
  top_pool_kernel<<<(ntot + 255) / 256, 256, 0, stream>>>(
      h, batch, top_g, top_b, top_m, top_v, g_sum, g_cnt, ntot);
  head_kernel<<<(BBC * OUTD + 255) / 256, 256, 0, stream>>>(
      g_sum, g_cnt, go_w, go_b, (float*)d_out);
}
